// ComminPub_18279380811971
// MI455X (gfx1250) — matrix operand order repaired and checked by reading, not compiled, not run on hardware
//
#include <hip/hip_runtime.h>

// Problem constants (from reference)
#define Cch   256                 // channels == codebook dim == NUM_CODES
#define Hh    128
#define Ww    256
#define Bb    4
#define HW    (Hh * Ww)           // 32768 pixels per (b, c) plane
#define NPIX  (Bb * HW)           // 131072 total pixels

// Tiling
#define PITCH         264         // padded LDS row pitch in f16 (breaks bank pattern)
#define WAVES_PER_WG  8
#define PIX_PER_WAVE  16
#define PIX_PER_WG    (WAVES_PER_WG * PIX_PER_WAVE)   // 128

typedef _Float16 v16h __attribute__((ext_vector_type(16)));
typedef _Float16 v8h  __attribute__((ext_vector_type(8)));
typedef float    v8f  __attribute__((ext_vector_type(8)));

// ---------------------------------------------------------------------------
// Kernel 1: inv_norm[n] = 1 / sum_c codebook[n,c]^2          (256 values)
// ---------------------------------------------------------------------------
__global__ void ck_invnorm(const float* __restrict__ cb, float* __restrict__ invn) {
    int n = threadIdx.x;
    float s = 0.f;
    #pragma unroll 8
    for (int c = 0; c < Cch; ++c) {
        float v = cb[n * Cch + c];
        s += v * v;
    }
    invn[n] = 1.0f / s;
}

// ---------------------------------------------------------------------------
// Kernel 2: M[i,j] = sum_n cb[n,i] * cb[n,j] * inv_norm[n]   (256x256, f16)
// M is symmetric, stored row-major f16 (so row-major == the A-matrix order).
// ---------------------------------------------------------------------------
__global__ void ck_build_M(const float* __restrict__ cb,
                           const float* __restrict__ invn,
                           _Float16* __restrict__ M) {
    int i = blockIdx.x;    // uniform within block -> scalar loads
    int j = threadIdx.x;   // coalesced across lanes
    float s = 0.f;
    #pragma unroll 4
    for (int n = 0; n < Cch; ++n) {
        s += cb[n * Cch + i] * cb[n * Cch + j] * invn[n];
    }
    M[i * Cch + j] = (_Float16)s;
}

// ---------------------------------------------------------------------------
// Kernel 3: out[b,c,hw] = sum_k M[c,k] * feature[b,k,hw]
// One wave: 16 pixels x all 256 channels (16 accumulator tiles of 16x16 f32).
// K loop: 8 chunks of 32 using v_wmma_f32_16x16x32_f16, f32 accumulate.
// ---------------------------------------------------------------------------
__global__ void __launch_bounds__(256)
ck_recon_gemm(const float* __restrict__ feature,
              const _Float16* __restrict__ M,
              float* __restrict__ out) {
    extern __shared__ _Float16 Ml[];   // [256][PITCH] f16, padded rows

    const int tid = threadIdx.x;

    // ---- Stage M (128KB f16) into LDS: thread t copies row t as 32 x 16B ----
    {
        const uint4* __restrict__ src = (const uint4*)(M + tid * Cch);   // 512B-aligned
        uint4* dst = (uint4*)(Ml + tid * PITCH);                         // 528B pitch, 16B-aligned
        #pragma unroll
        for (int j = 0; j < 32; ++j) dst[j] = src[j];
    }
    __syncthreads();

    const int lane   = tid & 31;
    const int wave   = tid >> 5;
    const int lane15 = lane & 15;
    const bool hiHalf = (lane >= 16);

    // Pixel tile for this wave (tiles never cross a batch boundary: HW % 16 == 0)
    const int p0  = blockIdx.x * PIX_PER_WG + wave * PIX_PER_WAVE;
    const int b   = p0 / HW;
    const int hw0 = p0 % HW;

    v8f acc[16];
    #pragma unroll
    for (int t = 0; t < 16; ++t) acc[t] = (v8f){};

    // B-matrix (32x16 f16, K x pixels) per ISA layout:
    //   lanes 0-15: pixel = p0+lane,   K = kb*32 + {0..7, 16..23}
    //   lanes 16-31: pixel = p0+lane-16, K = kb*32 + {8..15, 24..31}
    const float* fbase = feature + ((size_t)b * Cch) * HW + hw0 + lane15;
    const int bK = hiHalf ? 8 : 0;
    // A-matrix (16x32 f16) per ISA layout:
    //   lanes 0-15:  row = ct*16+lane,    K = {0..7, 16..23}
    //   lanes 16-31: row = ct*16+lane-16, K = {8..15, 24..31}
    const int aK = hiHalf ? 8 : 0;

    for (int kb = 0; kb < 8; ++kb) {
        // ---- Load + convert B fragment: 16 coalesced b32 loads, f32 -> f16 ----
        v16h bf;
        const float* fp = fbase + (size_t)(kb * 32 + bK) * HW;
        #pragma unroll
        for (int j = 0; j < 16; ++j) {
            bf[j] = (_Float16)fp[(size_t)(j < 8 ? j : j + 8) * HW];   // imm offsets: up to 23*HW*4 < 3MB, fits I24
        }

        const int kA = kb * 32 + aK;
        #pragma unroll
        for (int ct = 0; ct < 16; ++ct) {
            // ---- A fragment from LDS: two 16B chunks per lane ----
            const _Float16* ap = Ml + (ct * 16 + lane15) * PITCH + kA;
            v8h lo = *(const v8h*)ap;          // K = kA .. kA+7
            v8h hi = *(const v8h*)(ap + 16);   // K = kA+16 .. kA+23
            v16h af = __builtin_shufflevector(lo, hi,
                        0,1,2,3,4,5,6,7,8,9,10,11,12,13,14,15);

            acc[ct] = __builtin_amdgcn_wmma_f32_16x16x32_f16(
                          /*neg_a=*/false, af, /*neg_b=*/false, bf,
                          /*c_mod=*/(short)0, acc[ct],
                          /*reuse_a=*/false, /*reuse_b=*/false);
        }
    }

    // ---- Store: f32 C/D layout. VGPR r: lanes0-15 -> ch = ct*16+r,
    //      lanes16-31 -> ch = ct*16+r+8; pixel = hw0 + lane15. ----
    float* obase = out + ((size_t)b * Cch) * HW + hw0 + lane15;
    const int cOff = hiHalf ? 8 : 0;
    #pragma unroll
    for (int ct = 0; ct < 16; ++ct) {
        #pragma unroll
        for (int r = 0; r < 8; ++r) {
            obase[(size_t)(ct * 16 + r + cOff) * HW] = acc[ct][r];
        }
    }
}

// ---------------------------------------------------------------------------
extern "C" void kernel_launch(void* const* d_in, const int* in_sizes, int n_in,
                              void* d_out, int out_size, void* d_ws, size_t ws_size,
                              hipStream_t stream) {
    const float* feature  = (const float*)d_in[0];   // [B, C, H, W] f32
    const float* codebook = (const float*)d_in[1];   // [N, C] f32
    float* out = (float*)d_out;                      // [B, C, H, W] f32

    // Workspace: [0,1KB) inv_norm f32[256]; [1KB, 1KB+128KB) M f16[256*256]
    float*    invn = (float*)d_ws;
    _Float16* M    = (_Float16*)((char*)d_ws + 1024);

    ck_invnorm<<<1, 256, 0, stream>>>(codebook, invn);
    ck_build_M<<<Cch, 256, 0, stream>>>(codebook, invn, M);

    const size_t ldsBytes = (size_t)Cch * PITCH * sizeof(_Float16);   // 132 KB < 320 KB
    ck_recon_gemm<<<NPIX / PIX_PER_WG, WAVES_PER_WG * 32, ldsBytes, stream>>>(
        feature, M, out);
}
